// DensityDecoder_69879117906021
// MI455X (gfx1250) — matrix operand order repaired and checked by reading, not compiled, not run on hardware
//
#include <hip/hip_runtime.h>
#include <cmath>

// ---------------------------------------------------------------------------
// DensityDecoder on MI455X (gfx1250).
//  - All 1x1-conv GEMMs on v_wmma_f32_16x16x32_f16 (f32 accumulate).
//  - 64x64 per-wave macro-tile (16 accumulators) => ~128 FLOP per L2 byte,
//    compute-bound against the WMMA ceiling instead of L2-bound.
//  - A tiles staged into double-buffered LDS via GLOBAL_LOAD_ASYNC_TO_LDS_B128
//    (ASYNCcnt) — builtin confirmed present; args are (v4i AS1*, v4i AS3*, Ii, Ii).
//  - Neighborhood attention (clamped 7^3 window, head dims 84/112) on VALU.
// ---------------------------------------------------------------------------

typedef __attribute__((ext_vector_type(16))) _Float16 v16h;
typedef __attribute__((ext_vector_type(8)))  _Float16 v8h_t;
typedef __attribute__((ext_vector_type(8)))  float    v8f;
typedef int v4i_vec __attribute__((vector_size(16)));

#if defined(__gfx1250__) && __has_builtin(__builtin_amdgcn_global_load_async_to_lds_b128)
#  define USE_ASYNC_LDS 1
typedef __attribute__((address_space(1))) v4i_vec* gptr_b128;
typedef __attribute__((address_space(3))) v4i_vec* lptr_b128;
#else
#  define USE_ASYNC_LDS 0
#endif

static __device__ __forceinline__ int iclamp(int x, int lo, int hi) {
    return x < lo ? lo : (x > hi ? hi : x);
}

// ---------------------------------------------------------------------------
// f32 -> f16 conversion (weights + activations, once per GEMM)
// ---------------------------------------------------------------------------
__global__ void cvt_f16_kernel(const float* __restrict__ src,
                               _Float16* __restrict__ dst, int n) {
    int i = blockIdx.x * blockDim.x + threadIdx.x;
    if (i < n) dst[i] = (_Float16)src[i];
}

// ---------------------------------------------------------------------------
// WMMA GEMM: C[M,N] = Ah[M,K] @ Bh[N,K]^T + bias (+gelu) (+residual)
//  - 256 threads = 8 waves; block owns a 64-row M tile (shared LDS A stage)
//  - each wave computes a 64(M) x 64(N) macro-tile: 4x4 WMMA accumulators
//  - M%64==0, N%64==0, K%32==0 hold for every layer in this network
// ---------------------------------------------------------------------------
__global__ __launch_bounds__(256)
void gemm_wmma_kernel(const _Float16* __restrict__ Ah, const _Float16* __restrict__ Bh,
                      const float* __restrict__ bias, const float* __restrict__ res,
                      float* __restrict__ C, int M, int N, int K, int act) {
    __shared__ __align__(32) _Float16 Atile[2][64 * 32];   // 4KB per buffer

    const int tid  = threadIdx.x;
    const int wave = tid >> 5;
    const int lane = tid & 31;

    const int ntiles = N >> 6;
    int nt = blockIdx.x * 8 + wave;
    if (nt >= ntiles) nt = ntiles - 1;           // uniform clamp (EXEC stays full)
    const int mt = blockIdx.y;
    const int n0 = nt << 6;
    const int m0 = mt << 6;

    v8f acc[4][4];
#pragma unroll
    for (int ms = 0; ms < 4; ++ms)
#pragma unroll
        for (int ns = 0; ns < 4; ++ns)
            acc[ms][ns] = (v8f){0.f,0.f,0.f,0.f,0.f,0.f,0.f,0.f};

    const int row   = lane & 15;                 // M row within frag / N col
    const int abase = (lane < 16) ? 0 : 16;      // A frag K-half select
    const int bbase = (lane < 16) ? 0 : 16;      // B frag K-half select

    // staging: 64x32 f16 tile, 16B per thread
    const int arow = tid >> 2;                   // 0..63
    const int acol = (tid & 3) << 3;             // 0,8,16,24
    const int ksteps = K >> 5;

    for (int kt = 0; kt < ksteps; ++kt) {
        _Float16* buf = Atile[kt & 1];
        const int kbase = kt << 5;
        {
            const _Float16* gsrc = &Ah[(size_t)(m0 + arow) * K + kbase + acol];
            _Float16* ldst = &buf[arow * 32 + acol];
#if USE_ASYNC_LDS
            __builtin_amdgcn_global_load_async_to_lds_b128(
                (gptr_b128)gsrc, (lptr_b128)ldst, 0, 0);
#if __has_builtin(__builtin_amdgcn_s_wait_asynccnt)
            __builtin_amdgcn_s_wait_asynccnt(0);
#else
            asm volatile("s_wait_asynccnt 0x0" ::: "memory");
#endif
#else
            *(v8h_t*)ldst = *(const v8h_t*)gsrc;
#endif
        }
        __syncthreads();

        // B fragments: 32 contiguous bytes per lane, lanes 0-15 K0..15 / 16-31 K16..31
        const int kb = kbase + bbase;
        v16h bfrag[4];
#pragma unroll
        for (int ns = 0; ns < 4; ++ns)
            bfrag[ns] = *(const v16h*)&Bh[(size_t)(n0 + (ns << 4) + row) * K + kb];

#pragma unroll
        for (int ms = 0; ms < 4; ++ms) {
            // A fragment: lanes 0-15 -> K 0..15, lanes 16-31 -> 16..31
            const _Float16* ap = &buf[((ms << 4) + row) * 32 + abase];
            v8h_t alo = *(const v8h_t*)ap;
            v8h_t ahi = *(const v8h_t*)(ap + 8);
            v16h afrag;
#pragma unroll
            for (int i = 0; i < 8; ++i) { afrag[i] = alo[i]; afrag[8 + i] = ahi[i]; }
#pragma unroll
            for (int ns = 0; ns < 4; ++ns)
                acc[ms][ns] = __builtin_amdgcn_wmma_f32_16x16x32_f16(
                    false, afrag, false, bfrag[ns], (short)0, acc[ms][ns], false, false);
        }
        __syncthreads();
    }

    // epilogue: D layout — lane%16 = N col; VGPR r = M row r + 8*(lane/16)
    const int mrow = (lane >> 4) << 3;
#pragma unroll
    for (int ms = 0; ms < 4; ++ms) {
#pragma unroll
        for (int ns = 0; ns < 4; ++ns) {
            const int n = n0 + (ns << 4) + row;
            const float bv = bias[n];
#pragma unroll
            for (int r = 0; r < 8; ++r) {
                const int m = m0 + (ms << 4) + mrow + r;
                float v = acc[ms][ns][r] + bv;
                if (act == 1) v = 0.5f * v * (1.0f + erff(v * 0.70710678118654752f));
                if (res) v += res[(size_t)m * N + n];
                C[(size_t)m * N + n] = v;
            }
        }
    }
}

// ---------------------------------------------------------------------------
// AdaRMSNorm: per-channel spatial stats over token-major [V, C]
// ---------------------------------------------------------------------------
__global__ void stats_kernel(const float* __restrict__ X, int V, int C,
                             float* __restrict__ rms, float* __restrict__ mu) {
    int c = blockIdx.x * blockDim.x + threadIdx.x;
    if (c >= C) return;
    float s = 0.f, s2 = 0.f;
    for (int v = 0; v < V; ++v) {
        float x = X[(size_t)v * C + c];
        s += x; s2 += x * x;
    }
    float inv = 1.0f / (float)V;
    rms[c] = sqrtf(s2 * inv + 1e-6f);
    mu[c]  = s * inv;
}

__global__ __launch_bounds__(256)
void gate_kernel(const float* __restrict__ mu, const float* __restrict__ rms,
                 const float* __restrict__ scale,
                 const float* __restrict__ w1, const float* __restrict__ b1,
                 const float* __restrict__ w2, const float* __restrict__ b2,
                 int C, float* __restrict__ sfac) {
    __shared__ float hidden[2688];               // 2*C, C <= 1344
    const int t = threadIdx.x;
    const int C2 = 2 * C;
    for (int j = t; j < C2; j += 256) {
        float a = b1[j];
        for (int c = 0; c < C; ++c) a += w1[(size_t)j * C + c] * mu[c];
        hidden[j] = fmaxf(a, 0.f);
    }
    __syncthreads();
    for (int c = t; c < C; c += 256) {
        float g = b2[c];
        for (int j = 0; j < C2; ++j) g += w2[(size_t)c * C2 + j] * hidden[j];
        g = 1.0f / (1.0f + __expf(-g));
        sfac[c] = scale[c] * g / rms[c];
    }
}

__global__ void scalemul_kernel(const float* __restrict__ X,
                                const float* __restrict__ sfac,
                                float* __restrict__ Y, long long n, int C) {
    long long i = (long long)blockIdx.x * blockDim.x + threadIdx.x;
    if (i >= n) return;
    int c = (int)(i % C);
    Y[i] = X[i] * sfac[c];
}

// ---------------------------------------------------------------------------
// NeighborhoodAttention3D, stride=1, dilation=1, kernel 7 (clamped window).
// One block per (query token, head). qkv token-major [V, 3e]: q|k|v slabs.
// ---------------------------------------------------------------------------
__global__ __launch_bounds__(256)
void na3d_kernel(const float* __restrict__ qkv, float* __restrict__ out,
                 int V, int e, int heads, int hd, int D, float scale) {
    const int q = blockIdx.x / heads;
    const int h = blockIdx.x % heads;
    const int t = threadIdx.x;
    const int K3 = 343;

    __shared__ float sc[343];
    __shared__ float red[8];

    const int qd = q / (D * D), qh = (q / D) % D, qw = q % D;
    const int sd = iclamp(qd - 3, 0, D - 7);
    const int sh = iclamp(qh - 3, 0, D - 7);
    const int sw = iclamp(qw - 3, 0, D - 7);

    const size_t E3 = (size_t)3 * e;
    const int h0 = h * hd;
    const float* qrow = qkv + (size_t)q * E3 + h0;

    // scores over the exact clamped 7^3 window
    for (int j = t; j < K3; j += 256) {
        const int jd = j / 49, jh = (j / 7) % 7, jw = j % 7;
        const int key = ((sd + jd) * D + (sh + jh)) * D + (sw + jw);
        const float* krow = qkv + (size_t)key * E3 + e + h0;
        float s = 0.f;
        for (int c = 0; c < hd; ++c) s += qrow[c] * krow[c];
        sc[j] = s * scale;
    }
    __syncthreads();

    // block max
    float mx = -1e30f;
    for (int j = t; j < K3; j += 256) mx = fmaxf(mx, sc[j]);
    for (int off = 16; off > 0; off >>= 1) mx = fmaxf(mx, __shfl_xor(mx, off, 32));
    if ((t & 31) == 0) red[t >> 5] = mx;
    __syncthreads();
    if (t == 0) {
        float m2 = red[0];
        for (int i = 1; i < 8; ++i) m2 = fmaxf(m2, red[i]);
        red[0] = m2;
    }
    __syncthreads();
    mx = red[0];
    __syncthreads();

    // exp + block sum
    float ps = 0.f;
    for (int j = t; j < K3; j += 256) {
        float p = __expf(sc[j] - mx);
        sc[j] = p; ps += p;
    }
    for (int off = 16; off > 0; off >>= 1) ps += __shfl_xor(ps, off, 32);
    __syncthreads();
    if ((t & 31) == 0) red[t >> 5] = ps;
    __syncthreads();
    if (t == 0) {
        float s2 = 0.f;
        for (int i = 0; i < 8; ++i) s2 += red[i];
        red[0] = s2;
    }
    __syncthreads();
    const float inv = 1.0f / red[0];

    // weighted sum of V rows, one channel per thread (hd <= 112)
    for (int c = t; c < hd; c += 256) {
        float o = 0.f;
        for (int j = 0; j < K3; ++j) {
            const int jd = j / 49, jh = (j / 7) % 7, jw = j % 7;
            const int key = ((sd + jd) * D + (sh + jh)) * D + (sw + jw);
            o += sc[j] * qkv[(size_t)key * E3 + 2 * (size_t)e + h0 + c];
        }
        out[(size_t)q * e + h0 + c] = o * inv;
    }
}

// ---------------------------------------------------------------------------
// Layout shuffles
// ---------------------------------------------------------------------------
__global__ void transpose_in_kernel(const float* __restrict__ in,
                                    float* __restrict__ out) {
    // [C=512, V=512] channel-major -> token-major [V, C]
    int i = blockIdx.x * 256 + threadIdx.x;      // 262144
    int c = i >> 9, v = i & 511;
    out[(size_t)v * 512 + c] = in[i];
}

__global__ void ps2_kernel(const float* __restrict__ in, float* __restrict__ out) {
    // pixel_shuffle3d(f=2): [512 tok(8^3), 512 ch] -> [4096 tok(16^3), 64 ch]
    int i = blockIdx.x * 256 + threadIdx.x;      // 262144
    int c  = i & 63;
    int vo = i >> 6;
    int ow = vo & 15, oh = (vo >> 4) & 15, od = vo >> 8;
    int d = od >> 1, fd = od & 1;
    int h = oh >> 1, fh = oh & 1;
    int w = ow >> 1, fw = ow & 1;
    int cin = c * 8 + fd * 4 + fh * 2 + fw;
    int vi = (d * 8 + h) * 8 + w;
    out[i] = in[(size_t)vi * 512 + cin];
}

__global__ void ps4_out_kernel(const float* __restrict__ in,
                               const float* __restrict__ ocw,
                               const float* __restrict__ ocb,
                               float* __restrict__ out) {
    // pixel_shuffle3d(f=4): [4096 tok(16^3), 64 ch] -> [64^3] + 1x1 out_conv
    int i = blockIdx.x * 256 + threadIdx.x;      // 262144
    int ow = i & 63, oh = (i >> 6) & 63, od = i >> 12;
    int d = od >> 2, fd = od & 3;
    int h = oh >> 2, fh = oh & 3;
    int w = ow >> 2, fw = ow & 3;
    int c = fd * 16 + fh * 4 + fw;
    int vi = (d * 16 + h) * 16 + w;
    out[i] = in[(size_t)vi * 64 + c] * ocw[0] + ocb[0];
}

// ---------------------------------------------------------------------------
// Host orchestration
// ---------------------------------------------------------------------------
extern "C" void kernel_launch(void* const* d_in, const int* in_sizes, int n_in,
                              void* d_out, int out_size, void* d_ws, size_t ws_size,
                              hipStream_t stream) {
    (void)in_sizes; (void)n_in; (void)out_size; (void)ws_size;

    auto F = [&](int i) { return (const float*)d_in[i]; };

    // workspace carve-out (256B aligned slabs)
    char* ws = (char*)d_ws;
    size_t off = 0;
    auto alloc = [&](size_t bytes) -> char* {
        char* p = ws + off;
        off = (off + bytes + 255) & ~(size_t)255;
        return p;
    };
    float* X0 = (float*)alloc((size_t)262144 * 4);
    float* X1 = (float*)alloc((size_t)262144 * 4);
    const size_t EMAX = (size_t)4096 * 896;      // max V*e
    float* E1 = (float*)alloc(EMAX * 4);
    float* E2 = (float*)alloc(EMAX * 4);
    float* E3 = (float*)alloc(EMAX * 4);
    float* Qb = (float*)alloc((size_t)4096 * 2688 * 4);       // max V*3e
    float* Hb = (float*)alloc((size_t)4096 * 3584 * 4);       // max V*4e
    _Float16* WH = (_Float16*)alloc((size_t)5376 * 1344 * 2); // max O*K f16
    _Float16* AH = (_Float16*)alloc((size_t)4096 * 3584 * 2); // max M*K f16
    float* rms  = (float*)alloc(1344 * 4);
    float* mu   = (float*)alloc(1344 * 4);
    float* sfac = (float*)alloc(1344 * 4);

    auto gemm = [&](const float* A, const float* Wf, const float* bias,
                    const float* res, float* Cout, int M, int N, int K, int act) {
        const int nK = N * K;
        const int mK = M * K;
        cvt_f16_kernel<<<dim3((nK + 255) / 256), dim3(256), 0, stream>>>(Wf, WH, nK);
        cvt_f16_kernel<<<dim3((mK + 255) / 256), dim3(256), 0, stream>>>(A, AH, mK);
        dim3 g((unsigned)((N / 64 + 7) / 8), (unsigned)(M / 64));
        gemm_wmma_kernel<<<g, dim3(256), 0, stream>>>(AH, WH, bias, res, Cout, M, N, K, act);
    };

    auto rmsnorm = [&](const float* X, float* Y, int V, int C, const float* sc,
                       const float* w1, const float* b1,
                       const float* w2, const float* b2) {
        stats_kernel<<<dim3((C + 255) / 256), dim3(256), 0, stream>>>(X, V, C, rms, mu);
        gate_kernel<<<dim3(1), dim3(256), 0, stream>>>(mu, rms, sc, w1, b1, w2, b2, C, sfac);
        long long n = (long long)V * C;
        scalemul_kernel<<<dim3((unsigned)((n + 255) / 256)), dim3(256), 0, stream>>>(X, sfac, Y, n, C);
    };

    // param leaves, insertion order per block:
    // proj_in{w,b} norm1{scale,w1,b1,w2,b2} qkv{w,b} attn_proj{w,b}
    // norm2{scale,w1,b1,w2,b2} mlp_w1{w,b} mlp_w2{w,b} proj_out{w,b}  (22 leaves)
    auto run_block = [&](const float* Xin, float* Xout, int base,
                         int V, int cin, int e, int heads, int D) {
        const float *pin_w = F(base + 0),  *pin_b = F(base + 1);
        const float *n1_s  = F(base + 2),  *n1_w1 = F(base + 3), *n1_b1 = F(base + 4),
                    *n1_w2 = F(base + 5),  *n1_b2 = F(base + 6);
        const float *qkv_w = F(base + 7),  *qkv_b = F(base + 8);
        const float *ap_w  = F(base + 9),  *ap_b  = F(base + 10);
        const float *n2_s  = F(base + 11), *n2_w1 = F(base + 12), *n2_b1 = F(base + 13),
                    *n2_w2 = F(base + 14), *n2_b2 = F(base + 15);
        const float *m1_w  = F(base + 16), *m1_b  = F(base + 17);
        const float *m2_w  = F(base + 18), *m2_b  = F(base + 19);
        const float *po_w  = F(base + 20), *po_b  = F(base + 21);
        const int hd = e / heads;
        const float scl = 1.0f / sqrtf((float)hd);

        gemm(Xin, pin_w, pin_b, nullptr, E1, V, e, cin, 0);            // x = proj_in
        rmsnorm(E1, E2, V, e, n1_s, n1_w1, n1_b1, n1_w2, n1_b2);       // norm1
        gemm(E2, qkv_w, qkv_b, nullptr, Qb, V, 3 * e, e, 0);           // qkv
        na3d_kernel<<<dim3((unsigned)(V * heads)), dim3(256), 0, stream>>>(
            Qb, E2, V, e, heads, hd, D, scl);                          // attention
        gemm(E2, ap_w, ap_b, E1, E3, V, e, e, 0);                      // x += proj(o)
        rmsnorm(E3, E2, V, e, n2_s, n2_w1, n2_b1, n2_w2, n2_b2);       // norm2
        gemm(E2, m1_w, m1_b, nullptr, Hb, V, 4 * e, e, 1);             // gelu(mlp1)
        gemm(Hb, m2_w, m2_b, E3, E1, V, e, 4 * e, 0);                  // x += mlp2
        gemm(E1, po_w, po_b, nullptr, Xout, V, cin, e, 0);             // proj_out
    };

    // input x: [1,512,8,8,8] channel-major -> token-major [512,512]
    transpose_in_kernel<<<dim3(1024), dim3(256), 0, stream>>>(F(0), X0);

    run_block(X0, X1, 1,  512, 512, 1344, 16, 8);   // decode1[0]
    run_block(X1, X0, 23, 512, 512, 1344, 16, 8);   // decode1[1]

    ps2_kernel<<<dim3(1024), dim3(256), 0, stream>>>(X0, X1);  // -> [4096,64]

    run_block(X1, X0, 45, 4096, 64, 896, 8, 16);    // decode2[0]
    run_block(X0, X1, 67, 4096, 64, 896, 8, 16);    // decode2[1]

    // pixel_shuffle(4) + out_conv -> d_out [1,1,64,64,64] fp32
    ps4_out_kernel<<<dim3(1024), dim3(256), 0, stream>>>(
        X1, F(89), F(90), (float*)d_out);
}
